// TransactionGNN_2774548873485
// MI455X (gfx1250) — compile-verified, matrix operand order checked by reading, not run on hardware
//
#include <hip/hip_runtime.h>

// Problem constants (match reference)
#define NT 200000
#define NM 10000
#define NC 400
#define NE 400000
#define HD 64
#define NCLS 400

typedef __attribute__((ext_vector_type(16))) __bf16 v16bf;
typedef __attribute__((ext_vector_type(8)))  float  v8f;

enum { GF_ACC = 1, GF_RELU = 2 };

// Native fptrunc f32->bf16 (RTE): backend emits packed v_cvt_pk_bf16_f32.
__device__ __forceinline__ __bf16 f2bf(float f) { return (__bf16)f; }

// pack 8 consecutive-K floats (two float4) into elements [base, base+8) of an A/B fragment
__device__ __forceinline__ void pack8(v16bf& v, int base, float4 a, float4 b, float s) {
    v[base + 0] = f2bf(a.x * s); v[base + 1] = f2bf(a.y * s);
    v[base + 2] = f2bf(a.z * s); v[base + 3] = f2bf(a.w * s);
    v[base + 4] = f2bf(b.x * s); v[base + 5] = f2bf(b.y * s);
    v[base + 6] = f2bf(b.z * s); v[base + 7] = f2bf(b.w * s);
}

// Load one 16x32 A fragment for K-chunk kk of a row-major [*, K] matrix.
// Lane (g, hr) holds row m0+hr; elements 0..7 = K kk+8g..+7, 8..15 = K kk+16+8g..+7.
template<int K>
__device__ __forceinline__ v16bf load_a_frag(const float* __restrict__ A, int row, int kk,
                                             int g, float s) {
    v16bf a;
    const float* ap = A + (size_t)row * K + kk + 8 * g;
    float4 x0 = *reinterpret_cast<const float4*>(ap);
    float4 x1 = *reinterpret_cast<const float4*>(ap + 4);
    pack8(a, 0, x0, x1, s);
    if (kk + 24 <= K) {                        // compile-time for K in {16,32,64}
        float4 y0 = *reinterpret_cast<const float4*>(ap + 16);
        float4 y1 = *reinterpret_cast<const float4*>(ap + 20);
        pack8(a, 8, y0, y1, s);
    } else {
#pragma unroll
        for (int e = 8; e < 16; ++e) a[e] = f2bf(0.0f);
    }
    return a;
}

// Load one 32x16 B fragment (K-chunk kk) of row-major W[K x N]; lane holds column bcol.
template<int K>
__device__ __forceinline__ v16bf load_b_frag(const float* __restrict__ W, int bcol, int kk,
                                             int g, int N) {
    v16bf b;
#pragma unroll
    for (int e = 0; e < 16; ++e) {
        int k = kk + ((e < 8) ? e : (e + 8)) + (g << 3);
        b[e] = (k < K) ? f2bf(W[(size_t)k * N + bcol]) : f2bf(0.0f);  // folds per-element
    }
    return b;
}

// ---------------- single-source GEMM (encoders / classifier) ----------------
// C[M x N] = A[M x K] @ W[K x N] + bias.  M % 16 == 0. 4 waves/block = 16x64 cols,
// each wave iterates MT row tiles reusing its B fragments.
template<int K, int MT>
__global__ void __launch_bounds__(128)
gemm1_kernel(const float* __restrict__ A, const float* __restrict__ W,
             const float* __restrict__ bias, float* __restrict__ C, int M, int N)
{
    const int lane  = threadIdx.x & 31;
    const int wave  = threadIdx.x >> 5;
    const int nTile = blockIdx.y * 4 + wave;
    if (nTile * 16 >= N) return;               // wave-uniform
    const int g  = lane >> 4;
    const int hr = lane & 15;
    const int n0 = nTile * 16;

    constexpr int NF = (K + 31) / 32;
    v16bf bfrag[NF];
#pragma unroll
    for (int f = 0; f < NF; ++f) bfrag[f] = load_b_frag<K>(W, n0 + hr, f * 32, g, N);

    float bv = bias[n0 + hr];                  // bias for this lane's epilogue column

    for (int mt = 0; mt < MT; ++mt) {
        int tile = blockIdx.x * MT + mt;
        if (tile * 16 >= M) break;             // wave-uniform
        int m0 = tile * 16;
        v8f acc = {};
#pragma unroll
        for (int f = 0; f < NF; ++f) {
            v16bf a = load_a_frag<K>(A, m0 + hr, f * 32, g, 1.0f);
            acc = __builtin_amdgcn_wmma_f32_16x16x32_bf16(false, a, false, bfrag[f],
                                                          (short)0, acc, false, false);
        }
#pragma unroll
        for (int i = 0; i < 8; ++i) {
            int row = m0 + i + (g << 3);
            C[(size_t)row * N + n0 + hr] = acc[i] + bv;
        }
    }
}

// ---------------- fused SAGE conv GEMM ----------------
// C[M x 64] = op( (A0 * 1/max(cnt0,1)) @ W0 + A1 @ W1 + bias (+ C) ), op = optional relu.
// A0, A1 are [M x 64]; W0, W1 are [64 x 64]. N fixed to HD=64, one n-tile per wave.
// FLAGS is compile-time: epilogue is branch-free.
template<int MT, int FLAGS>
__global__ void __launch_bounds__(128)
conv_gemm_kernel(const float* __restrict__ A0, const float* __restrict__ cnt0,
                 const float* __restrict__ W0,
                 const float* __restrict__ A1, const float* __restrict__ W1,
                 const float* __restrict__ bias, float* __restrict__ C, int M)
{
    const int lane = threadIdx.x & 31;
    const int wave = threadIdx.x >> 5;
    const int g    = lane >> 4;
    const int hr   = lane & 15;
    const int n0   = wave * 16;

    v16bf bfrag[4];
#pragma unroll
    for (int f = 0; f < 2; ++f) bfrag[f]     = load_b_frag<HD>(W0, n0 + hr, f * 32, g, HD);
#pragma unroll
    for (int f = 0; f < 2; ++f) bfrag[2 + f] = load_b_frag<HD>(W1, n0 + hr, f * 32, g, HD);

    float bv = bias[n0 + hr];

    for (int mt = 0; mt < MT; ++mt) {
        int tile = blockIdx.x * MT + mt;
        if (tile * 16 >= M) break;             // wave-uniform
        int m0  = tile * 16;
        int row = m0 + hr;
        float s0 = 1.0f / fmaxf(cnt0[row], 1.0f);   // mean normalization folded in

        v8f acc = {};
#pragma unroll
        for (int f = 0; f < 2; ++f) {
            v16bf a = load_a_frag<HD>(A0, row, f * 32, g, s0);
            acc = __builtin_amdgcn_wmma_f32_16x16x32_bf16(false, a, false, bfrag[f],
                                                          (short)0, acc, false, false);
        }
#pragma unroll
        for (int f = 0; f < 2; ++f) {
            v16bf a = load_a_frag<HD>(A1, row, f * 32, g, 1.0f);
            acc = __builtin_amdgcn_wmma_f32_16x16x32_bf16(false, a, false, bfrag[2 + f],
                                                          (short)0, acc, false, false);
        }

        float prev[8];
        if constexpr (FLAGS & GF_ACC) {        // compile-time: loads can clause
#pragma unroll
            for (int i = 0; i < 8; ++i) {
                int r = m0 + i + (g << 3);
                prev[i] = C[(size_t)r * HD + n0 + hr];
            }
        }
#pragma unroll
        for (int i = 0; i < 8; ++i) {
            int r = m0 + i + (g << 3);
            float v = acc[i] + bv;
            if constexpr (FLAGS & GF_ACC)  v += prev[i];
            if constexpr (FLAGS & GF_RELU) v  = fmaxf(v, 0.0f);
            C[(size_t)r * HD + n0 + hr] = v;
        }
    }
}

// agg[dst[e]] += feat[src[e]]   (64 features; 16 threads/edge, float4 per thread)
__global__ void __launch_bounds__(256)
scatter_add_kernel(const float* __restrict__ feat, const int* __restrict__ sidx,
                   const int* __restrict__ didx, float* __restrict__ agg, int E)
{
    long long t = (long long)blockIdx.x * blockDim.x + threadIdx.x;
    int e = (int)(t >> 4);
    if (e >= E) return;
    int part = (int)(t & 15);
    int s = sidx[e], d = didx[e];
    const float4 v = *reinterpret_cast<const float4*>(feat + (long long)s * HD + part * 4);
    float* o = agg + (long long)d * HD + part * 4;
    atomicAdd(o + 0, v.x);
    atomicAdd(o + 1, v.y);
    atomicAdd(o + 2, v.z);
    atomicAdd(o + 3, v.w);
}

__global__ void __launch_bounds__(256)
degree_kernel(const int* __restrict__ idx, float* __restrict__ deg, int E)
{
    int e = blockIdx.x * blockDim.x + threadIdx.x;
    if (e < E) atomicAdd(&deg[idx[e]], 1.0f);
}

extern "C" void kernel_launch(void* const* d_in, const int* in_sizes, int n_in,
                              void* d_out, int out_size, void* d_ws, size_t ws_size,
                              hipStream_t stream)
{
    (void)in_sizes; (void)n_in; (void)out_size; (void)ws_size;
    constexpr int MT = 4;

    const float* x_t      = (const float*)d_in[0];
    const float* x_m      = (const float*)d_in[1];
    const float* x_c      = (const float*)d_in[2];
    const int*   e_bt_src = (const int*)d_in[3];
    const int*   e_bt_dst = (const int*)d_in[4];
    const int*   e_tc_src = (const int*)d_in[5];
    const int*   e_tc_dst = (const int*)d_in[6];

    const float* enc_t_W = (const float*)d_in[7];
    const float* enc_t_b = (const float*)d_in[8];
    const float* enc_m_W = (const float*)d_in[9];
    const float* enc_m_b = (const float*)d_in[10];
    const float* enc_c_W = (const float*)d_in[11];
    const float* enc_c_b = (const float*)d_in[12];
    const float* cls_W   = (const float*)d_in[37];
    const float* cls_b   = (const float*)d_in[38];

    // workspace carve-out (floats); region sizes are multiples of 4 floats (16B)
    float* p = (float*)d_ws;
    auto take = [&](size_t nflt) { float* r = p; p += nflt; return r; };
    float* hA_t  = take((size_t)NT * HD);
    float* hB_t  = take((size_t)NT * HD);
    float* agg_t = take((size_t)NT * HD);
    float* hA_m  = take((size_t)NM * HD);
    float* hB_m  = take((size_t)NM * HD);
    float* agg_m = take((size_t)NM * HD);
    float* hA_c  = take((size_t)NC * HD);
    float* hB_c  = take((size_t)NC * HD);
    float* agg_c = take((size_t)NC * HD);
    float* deg_m    = take(NM);      // four degree arrays contiguous: one memset
    float* deg_c    = take(NC);
    float* deg_t_bt = take(NT);
    float* deg_t_tc = take(NT);

    auto scatter = [&](const float* feat, const int* s, const int* d, float* agg, int n_dst) {
        hipMemsetAsync(agg, 0, (size_t)n_dst * HD * sizeof(float), stream);
        long long T = (long long)NE * 16;
        scatter_add_kernel<<<(int)((T + 255) / 256), 256, 0, stream>>>(feat, s, d, agg, NE);
    };
    auto convGrid = [&](int M) { return dim3((M / 16 + MT - 1) / MT); };

    // ---- degrees (fixed across layers) ----
    hipMemsetAsync(deg_m, 0, (size_t)(NM + NC + NT + NT) * sizeof(float), stream);
    int degBlocks = (NE + 255) / 256;
    degree_kernel<<<degBlocks, 256, 0, stream>>>(e_bt_dst, deg_m,    NE);
    degree_kernel<<<degBlocks, 256, 0, stream>>>(e_tc_dst, deg_c,    NE);
    degree_kernel<<<degBlocks, 256, 0, stream>>>(e_bt_src, deg_t_bt, NE);
    degree_kernel<<<degBlocks, 256, 0, stream>>>(e_tc_src, deg_t_tc, NE);

    // ---- node encoders ----
    gemm1_kernel<64, MT><<<dim3((NT/16 + MT-1)/MT, 1), 128, 0, stream>>>(x_t, enc_t_W, enc_t_b, hA_t, NT, HD);
    gemm1_kernel<32, MT><<<dim3((NM/16 + MT-1)/MT, 1), 128, 0, stream>>>(x_m, enc_m_W, enc_m_b, hA_m, NM, HD);
    gemm1_kernel<16, MT><<<dim3((NC/16 + MT-1)/MT, 1), 128, 0, stream>>>(x_c, enc_c_W, enc_c_b, hA_c, NC, HD);

    float* h_t = hA_t; float* h_t_n = hB_t;
    float* h_m = hA_m; float* h_m_n = hB_m;
    float* h_c = hA_c; float* h_c_n = hB_c;

    for (int l = 0; l < 2; ++l) {
        int base = 13 + l * 12; // (W_l,b_l,W_r) x [bt, tc, rev_bt, rev_tc]
        const float* bt_Wl  = (const float*)d_in[base + 0];
        const float* bt_bl  = (const float*)d_in[base + 1];
        const float* bt_Wr  = (const float*)d_in[base + 2];
        const float* tc_Wl  = (const float*)d_in[base + 3];
        const float* tc_bl  = (const float*)d_in[base + 4];
        const float* tc_Wr  = (const float*)d_in[base + 5];
        const float* rbt_Wl = (const float*)d_in[base + 6];
        const float* rbt_bl = (const float*)d_in[base + 7];
        const float* rbt_Wr = (const float*)d_in[base + 8];
        const float* rtc_Wl = (const float*)d_in[base + 9];
        const float* rtc_bl = (const float*)d_in[base + 10];
        const float* rtc_Wr = (const float*)d_in[base + 11];

        // m_new = relu( mean_bt @ Wl + bl + h_m @ Wr )       (single fused GEMM)
        scatter(h_t, e_bt_src, e_bt_dst, agg_m, NM);
        conv_gemm_kernel<MT, GF_RELU><<<convGrid(NM), 128, 0, stream>>>(
            agg_m, deg_m, bt_Wl, h_m, bt_Wr, bt_bl, h_m_n, NM);

        // c_new = relu( mean_tc @ Wl + bl + h_c @ Wr )
        scatter(h_t, e_tc_src, e_tc_dst, agg_c, NC);
        conv_gemm_kernel<MT, GF_RELU><<<convGrid(NC), 128, 0, stream>>>(
            agg_c, deg_c, tc_Wl, h_c, tc_Wr, tc_bl, h_c_n, NC);

        // t_new = relu( sage_rev_bt + sage_rev_tc ), agg buffer reused sequentially
        scatter(h_m, e_bt_dst, e_bt_src, agg_t, NT);
        conv_gemm_kernel<MT, 0><<<convGrid(NT), 128, 0, stream>>>(
            agg_t, deg_t_bt, rbt_Wl, h_t, rbt_Wr, rbt_bl, h_t_n, NT);
        scatter(h_c, e_tc_dst, e_tc_src, agg_t, NT);
        conv_gemm_kernel<MT, GF_ACC | GF_RELU><<<convGrid(NT), 128, 0, stream>>>(
            agg_t, deg_t_tc, rtc_Wl, h_t, rtc_Wr, rtc_bl, h_t_n, NT);

        // ping-pong
        float* tmp;
        tmp = h_t; h_t = h_t_n; h_t_n = tmp;
        tmp = h_m; h_m = h_m_n; h_m_n = tmp;
        tmp = h_c; h_c = h_c_n; h_c_n = tmp;
    }

    // ---- classifier: [NT,64] @ [64,400] + b ----
    gemm1_kernel<64, MT><<<dim3((NT/16 + MT-1)/MT, (NCLS + 63)/64), 128, 0, stream>>>(
        h_t, cls_W, cls_b, (float*)d_out, NT, NCLS);
}
